// MultiHeadAttention_4329327035164
// MI455X (gfx1250) — compile-verified, matrix operand order checked by reading, not run on hardware
//
#include <hip/hip_runtime.h>

// ---------------- problem constants ----------------
#define EMBED   1024
#define HEADS   16
#define HDIM    64
#define SEQ     2048
#define NBATCH  4
#define MTOTAL  (NBATCH * SEQ)       // 8192 rows for the big GEMMs
#define SCALE   0.125f               // 1/sqrt(64)

// ---------------- vector types (trivial ext-vectors; safe in unions) ---------
typedef __attribute__((ext_vector_type(16))) __bf16   v16bf;
typedef __attribute__((ext_vector_type(2)))  __bf16   bf16x2;
typedef __attribute__((ext_vector_type(8)))  float    v8f;
typedef __attribute__((ext_vector_type(4)))  unsigned u32x4;
typedef __attribute__((ext_vector_type(2)))  unsigned u32x2;
typedef __attribute__((ext_vector_type(4)))  float    f32x4;

union Frag { v16bf v; u32x4 u[2]; };   // 32B: one WMMA A/B operand per lane

static __device__ __forceinline__ v8f vzero() {
  v8f z;
  #pragma unroll
  for (int i = 0; i < 8; ++i) z[i] = 0.0f;
  return z;
}

// native bf16 conversion (backend picks hardware cvt; RNE)
static __device__ __forceinline__ unsigned short f2bf(float f) {
  union { __bf16 h; unsigned short s; } u;
  u.h = (__bf16)f;
  return u.s;
}
static __device__ __forceinline__ unsigned pack_bf16x2(float a, float b) {
  union { bf16x2 v; unsigned u; } p;
  p.v[0] = (__bf16)a; p.v[1] = (__bf16)b;
  return p.u;
}

static __device__ __forceinline__ v8f wmma_bf16(const Frag& a, const Frag& b, v8f c) {
  return __builtin_amdgcn_wmma_f32_16x16x32_bf16(false, a.v, false, b.v,
                                                 (short)0, c, false, false);
}

// ---------------- gfx1250 async global->LDS copy (16B) ----------------------
static __device__ __forceinline__ void cp16_g2l(void* lds, const void* g) {
  unsigned           loff = (unsigned)(unsigned long long)lds;  // LDS addr = low 32 bits
  unsigned long long ga   = (unsigned long long)g;
  asm volatile("global_load_async_to_lds_b128 %0, %1, off"
               :: "v"(loff), "v"(ga) : "memory");
}
static __device__ __forceinline__ void wait_async0() {
  asm volatile("s_wait_asynccnt 0" ::: "memory");
}
static __device__ __forceinline__ void wait_async4() {   // allow 1 chunk (4 ops) in flight
  asm volatile("s_wait_asynccnt 4" ::: "memory");
}

// ============================================================================
// Kernel 1: Q/K/V projections.  C[m,n] = sum_k A[m,k] * W[n,k] + b[n]
// grid = (8, 64, 3); block = 256 (8 waves). 128x128 tile, K-step 32,
// register-pipelined global loads so HBM latency overlaps the WMMAs.
//   mode 0 -> Q  stored [b,h,s,d]   (bf16)
//   mode 1 -> K  stored [b,h,s,d]   (bf16)
//   mode 2 -> V  stored [b,h,d,s]   (bf16, transposed for the PV GEMM)
// ============================================================================
#define P_LDA 40   // 32 + 8 pad (80B row stride, 16B aligned)

__global__ __launch_bounds__(256) void proj_gemm(
    const float* __restrict__ q_in, const float* __restrict__ k_in,
    const float* __restrict__ v_in,
    const float* __restrict__ Wq, const float* __restrict__ Wk,
    const float* __restrict__ Wv,
    const float* __restrict__ bq, const float* __restrict__ bk,
    const float* __restrict__ bv,
    unsigned short* __restrict__ qws, unsigned short* __restrict__ kws,
    unsigned short* __restrict__ vtws)
{
  __shared__ unsigned short sA[128 * P_LDA];
  __shared__ unsigned short sB[128 * P_LDA];

  const int mode = blockIdx.z;
  const float* A    = (mode == 0) ? q_in : (mode == 1) ? k_in : v_in;
  const float* W    = (mode == 0) ? Wq   : (mode == 1) ? Wk   : Wv;
  const float* bias = (mode == 0) ? bq   : (mode == 1) ? bk   : bv;
  unsigned short* dst = (mode == 0) ? qws : (mode == 1) ? kws : vtws;

  const int tid  = threadIdx.x;
  const int lane = tid & 31;
  const int wave = tid >> 5;
  const int wm   = wave & 3;     // 4 waves along M (32 rows each)
  const int wn   = wave >> 2;    // 2 waves along N (64 cols each)
  const int m0   = blockIdx.y * 128;
  const int n0   = blockIdx.x * 128;
  const int kh0  = (lane < 16) ? 0 : 8;

  v8f acc[2][4];
  #pragma unroll
  for (int mt = 0; mt < 2; ++mt)
    #pragma unroll
    for (int nt = 0; nt < 4; ++nt) acc[mt][nt] = vzero();

  // register pipeline: this thread's 4 A-quads + 4 W-quads for the next k-step
  f32x4 ra[4], rw[4];
  #pragma unroll
  for (int j = 0; j < 4; ++j) {
    int i = tid + j * 256, row = i >> 3, kk = (i & 7) << 2;
    ra[j] = *(const f32x4*)(A + (size_t)(m0 + row) * EMBED + kk);
    rw[j] = *(const f32x4*)(W + (size_t)(n0 + row) * EMBED + kk);
  }

  for (int k0 = 0; k0 < EMBED; k0 += 32) {
    // convert + store the staged registers into LDS (f32 -> bf16)
    #pragma unroll
    for (int j = 0; j < 4; ++j) {
      int i = tid + j * 256, row = i >> 3, kk = (i & 7) << 2;
      u32x2 p; p[0] = pack_bf16x2(ra[j][0], ra[j][1]); p[1] = pack_bf16x2(ra[j][2], ra[j][3]);
      *(u32x2*)&sA[row * P_LDA + kk] = p;
      u32x2 q; q[0] = pack_bf16x2(rw[j][0], rw[j][1]); q[1] = pack_bf16x2(rw[j][2], rw[j][3]);
      *(u32x2*)&sB[row * P_LDA + kk] = q;
    }
    __syncthreads();

    // issue next k-step's global loads; they overlap the WMMAs below
    if (k0 + 32 < EMBED) {
      #pragma unroll
      for (int j = 0; j < 4; ++j) {
        int i = tid + j * 256, row = i >> 3, kk = (i & 7) << 2;
        ra[j] = *(const f32x4*)(A + (size_t)(m0 + row) * EMBED + k0 + 32 + kk);
        rw[j] = *(const f32x4*)(W + (size_t)(n0 + row) * EMBED + k0 + 32 + kk);
      }
    }

    Frag a[2], b[4];
    #pragma unroll
    for (int mt = 0; mt < 2; ++mt) {
      int row = wm * 32 + mt * 16 + (lane & 15);
      a[mt].u[0] = *(const u32x4*)&sA[row * P_LDA + kh0];
      a[mt].u[1] = *(const u32x4*)&sA[row * P_LDA + kh0 + 16];
    }
    #pragma unroll
    for (int nt = 0; nt < 4; ++nt) {
      int col = wn * 64 + nt * 16 + (lane & 15);
      b[nt].u[0] = *(const u32x4*)&sB[col * P_LDA + kh0];
      b[nt].u[1] = *(const u32x4*)&sB[col * P_LDA + kh0 + 16];
    }
    #pragma unroll
    for (int mt = 0; mt < 2; ++mt)
      #pragma unroll
      for (int nt = 0; nt < 4; ++nt)
        acc[mt][nt] = wmma_bf16(a[mt], b[nt], acc[mt][nt]);
    __syncthreads();
  }

  // epilogue: bias + scatter to per-head layout
  #pragma unroll
  for (int mt = 0; mt < 2; ++mt) {
    int rbase = m0 + wm * 32 + mt * 16 + ((lane >= 16) ? 8 : 0);
    #pragma unroll
    for (int nt = 0; nt < 4; ++nt) {
      int n = n0 + wn * 64 + nt * 16 + (lane & 15);
      float bb = bias[n];
      int h = n >> 6, d = n & 63;
      #pragma unroll
      for (int r = 0; r < 8; ++r) {
        int m = rbase + r;
        int bidx = m >> 11;          // / 2048
        int s    = m & 2047;
        unsigned short o = f2bf(acc[mt][nt][r] + bb);
        if (mode < 2)
          dst[(((size_t)(bidx * HEADS + h)) * SEQ + s) * HDIM + d] = o;
        else
          dst[(((size_t)(bidx * HEADS + h)) * HDIM + d) * SEQ + s] = o;
      }
    }
  }
}

// ============================================================================
// Kernel 2: flash attention.  grid = (S/256, H, N); block = 256 (8 waves).
// Each wave owns 32 query rows.  K/V chunks (64 keys) stream through
// double-buffered LDS via the gfx1250 async engine: chunk i+2 is issued as
// soon as buffer i%2 is free, and we wait with s_wait_asynccnt 4 (each wave
// issues exactly 4 async b128 ops per chunk; ASYNCcnt retires in order), so
// one chunk is always in flight behind the WMMA + softmax work.
// ============================================================================
#define A_LDK 72   // 64 + 8 pad (144B row stride)
#define NCHUNK (SEQ / 64)

__global__ __launch_bounds__(256) void attn_kernel(
    const unsigned short* __restrict__ qws, const unsigned short* __restrict__ kws,
    const unsigned short* __restrict__ vtws, unsigned short* __restrict__ attnws)
{
  __shared__ unsigned short sK[2][64 * A_LDK];        // K chunk  [kr][d]
  __shared__ unsigned short sV[2][64 * A_LDK];        // V^T chunk [d][kr]
  __shared__ unsigned short sP[8 * 32 * A_LDK];       // per-wave P transpose

  const int b    = blockIdx.z;
  const int h    = blockIdx.y;
  const int tid  = threadIdx.x;
  const int lane = tid & 31;
  const int wave = tid >> 5;
  const int qbase = blockIdx.x * 256 + wave * 32;
  const int kh0   = (lane < 16) ? 0 : 8;

  const unsigned short* Qp = qws  + ((size_t)(b * HEADS + h)) * SEQ * HDIM;
  const unsigned short* Kp = kws  + ((size_t)(b * HEADS + h)) * SEQ * HDIM;
  const unsigned short* Vp = vtws + ((size_t)(b * HEADS + h)) * HDIM * SEQ;
  unsigned short* myP = &sP[wave * 32 * A_LDK];

  // issue one K/V chunk's async copies (4 instructions per thread/wave)
  auto stage = [&](int kc, int buf) {
    #pragma unroll
    for (int j = 0; j < 2; ++j) {
      int i = tid + j * 256, row = i >> 3, dd = (i & 7) * 8;
      cp16_g2l(&sK[buf][row * A_LDK + dd], Kp + (size_t)(kc + row) * HDIM + dd);
    }
    #pragma unroll
    for (int j = 0; j < 2; ++j) {
      int i = tid + j * 256, dr = i >> 3, kk = (i & 7) * 8;
      cp16_g2l(&sV[buf][dr * A_LDK + kk], Vp + (size_t)dr * SEQ + kc + kk);
    }
  };

  // resident Q fragments: 2 M-tiles x 2 K-steps (Dh=64)
  Frag qf[2][2];
  #pragma unroll
  for (int mt = 0; mt < 2; ++mt) {
    int row = qbase + mt * 16 + (lane & 15);
    #pragma unroll
    for (int ks = 0; ks < 2; ++ks) {
      qf[mt][ks].u[0] = *(const u32x4*)(Qp + (size_t)row * HDIM + ks * 32 + kh0);
      qf[mt][ks].u[1] = *(const u32x4*)(Qp + (size_t)row * HDIM + ks * 32 + kh0 + 16);
    }
  }

  float mrow[2][8], lrow[2][8];
  v8f   o[2][4];
  #pragma unroll
  for (int mt = 0; mt < 2; ++mt) {
    #pragma unroll
    for (int r = 0; r < 8; ++r) { mrow[mt][r] = -3.0e38f; lrow[mt][r] = 0.0f; }
    #pragma unroll
    for (int dn = 0; dn < 4; ++dn) o[mt][dn] = vzero();
  }

  // prologue: two chunks in flight
  stage(0, 0);
  stage(64, 1);

  for (int ci = 0; ci < NCHUNK; ++ci) {
    const int buf = ci & 1;
    const unsigned short* cK = sK[buf];
    const unsigned short* cV = sV[buf];

    if (ci + 1 < NCHUNK) wait_async4(); else wait_async0();
    __syncthreads();

    // ---- scores: S = Q * K^T  (32 rows x 64 keys per wave) ----
    v8f sc[2][4];
    #pragma unroll
    for (int mt = 0; mt < 2; ++mt)
      #pragma unroll
      for (int nt = 0; nt < 4; ++nt) sc[mt][nt] = vzero();

    #pragma unroll
    for (int nt = 0; nt < 4; ++nt) {
      Frag bk0, bk1;
      int kr = nt * 16 + (lane & 15);
      bk0.u[0] = *(const u32x4*)&cK[kr * A_LDK + kh0];
      bk0.u[1] = *(const u32x4*)&cK[kr * A_LDK + kh0 + 16];
      bk1.u[0] = *(const u32x4*)&cK[kr * A_LDK + 32 + kh0];
      bk1.u[1] = *(const u32x4*)&cK[kr * A_LDK + 32 + kh0 + 16];
      #pragma unroll
      for (int mt = 0; mt < 2; ++mt) {
        sc[mt][nt] = wmma_bf16(qf[mt][0], bk0, sc[mt][nt]);
        sc[mt][nt] = wmma_bf16(qf[mt][1], bk1, sc[mt][nt]);
      }
    }

    // ---- online softmax (rows live in 16-lane halves; xor-reduce 1..8) ----
    #pragma unroll
    for (int mt = 0; mt < 2; ++mt) {
      #pragma unroll
      for (int r = 0; r < 8; ++r) {
        float mx = -3.0e38f;
        #pragma unroll
        for (int nt = 0; nt < 4; ++nt) mx = fmaxf(mx, sc[mt][nt][r] * SCALE);
        #pragma unroll
        for (int off = 1; off < 16; off <<= 1) mx = fmaxf(mx, __shfl_xor(mx, off, 32));
        float mnew  = fmaxf(mrow[mt][r], mx);
        float alpha = __expf(mrow[mt][r] - mnew);
        mrow[mt][r] = mnew;
        float psum = 0.0f;
        #pragma unroll
        for (int nt = 0; nt < 4; ++nt) {
          float p = __expf(sc[mt][nt][r] * SCALE - mnew);
          sc[mt][nt][r] = p;
          psum += p;
        }
        #pragma unroll
        for (int off = 1; off < 16; off <<= 1) psum += __shfl_xor(psum, off, 32);
        lrow[mt][r] = lrow[mt][r] * alpha + psum;
        #pragma unroll
        for (int dn = 0; dn < 4; ++dn) o[mt][dn][r] *= alpha;
      }
    }

    // ---- transpose P through per-wave LDS (C-layout -> A-layout) ----
    #pragma unroll
    for (int mt = 0; mt < 2; ++mt)
      #pragma unroll
      for (int nt = 0; nt < 4; ++nt) {
        int col = nt * 16 + (lane & 15);
        int rb  = mt * 16 + ((lane >= 16) ? 8 : 0);
        #pragma unroll
        for (int r = 0; r < 8; ++r)
          myP[(rb + r) * A_LDK + col] = f2bf(sc[mt][nt][r]);
      }

    // ---- O += P * V ----
    #pragma unroll
    for (int ks2 = 0; ks2 < 2; ++ks2) {
      Frag pf[2];
      #pragma unroll
      for (int mt = 0; mt < 2; ++mt) {
        int row = mt * 16 + (lane & 15);
        pf[mt].u[0] = *(const u32x4*)&myP[row * A_LDK + ks2 * 32 + kh0];
        pf[mt].u[1] = *(const u32x4*)&myP[row * A_LDK + ks2 * 32 + kh0 + 16];
      }
      #pragma unroll
      for (int dn = 0; dn < 4; ++dn) {
        Frag vf;
        int dcol = dn * 16 + (lane & 15);
        vf.u[0] = *(const u32x4*)&cV[dcol * A_LDK + ks2 * 32 + kh0];
        vf.u[1] = *(const u32x4*)&cV[dcol * A_LDK + ks2 * 32 + kh0 + 16];
        #pragma unroll
        for (int mt = 0; mt < 2; ++mt)
          o[mt][dn] = wmma_bf16(pf[mt], vf, o[mt][dn]);
      }
    }
    __syncthreads();

    // buffer `buf` is now free for everyone: refill it with chunk ci+2
    if (ci + 2 < NCHUNK) stage((ci + 2) * 64, buf);
  }

  // ---- epilogue: normalize and store bf16 [b, s, h*64+d] ----
  #pragma unroll
  for (int mt = 0; mt < 2; ++mt) {
    int rb = qbase + mt * 16 + ((lane >= 16) ? 8 : 0);
    #pragma unroll
    for (int dn = 0; dn < 4; ++dn) {
      int d = dn * 16 + (lane & 15);
      #pragma unroll
      for (int r = 0; r < 8; ++r) {
        float val = o[mt][dn][r] / lrow[mt][r];
        attnws[((size_t)(b * SEQ + rb + r)) * EMBED + h * HDIM + d] = f2bf(val);
      }
    }
  }
}

// ============================================================================
// Kernel 3: output projection.  out[m,n] = sum_k A_bf16[m,k]*Wo[n,k] + bo[n]
// grid = (8, 64); block = 256. A staged via async-LDS; Wo register-pipelined.
// ============================================================================
__global__ __launch_bounds__(256) void outproj_gemm(
    const unsigned short* __restrict__ A, const float* __restrict__ Wo,
    const float* __restrict__ bo, float* __restrict__ out)
{
  __shared__ unsigned short sA[128 * P_LDA];
  __shared__ unsigned short sB[128 * P_LDA];

  const int tid  = threadIdx.x;
  const int lane = tid & 31;
  const int wave = tid >> 5;
  const int wm   = wave & 3;
  const int wn   = wave >> 2;
  const int m0   = blockIdx.y * 128;
  const int n0   = blockIdx.x * 128;
  const int kh0  = (lane < 16) ? 0 : 8;

  v8f acc[2][4];
  #pragma unroll
  for (int mt = 0; mt < 2; ++mt)
    #pragma unroll
    for (int nt = 0; nt < 4; ++nt) acc[mt][nt] = vzero();

  f32x4 rw[4];
  #pragma unroll
  for (int j = 0; j < 4; ++j) {
    int i = tid + j * 256, row = i >> 3, kk = (i & 7) << 2;
    rw[j] = *(const f32x4*)(Wo + (size_t)(n0 + row) * EMBED + kk);
  }

  for (int k0 = 0; k0 < EMBED; k0 += 32) {
    // A is already bf16: async 16B copies straight into LDS
    #pragma unroll
    for (int j = 0; j < 2; ++j) {
      int i = tid + j * 256, row = i >> 2, kk = (i & 3) * 8;
      cp16_g2l(&sA[row * P_LDA + kk], A + (size_t)(m0 + row) * EMBED + k0 + kk);
    }
    // Wo: convert staged registers f32 -> bf16
    #pragma unroll
    for (int j = 0; j < 4; ++j) {
      int i = tid + j * 256, row = i >> 3, kk = (i & 7) << 2;
      u32x2 p; p[0] = pack_bf16x2(rw[j][0], rw[j][1]); p[1] = pack_bf16x2(rw[j][2], rw[j][3]);
      *(u32x2*)&sB[row * P_LDA + kk] = p;
    }
    wait_async0();
    __syncthreads();

    if (k0 + 32 < EMBED) {
      #pragma unroll
      for (int j = 0; j < 4; ++j) {
        int i = tid + j * 256, row = i >> 3, kk = (i & 7) << 2;
        rw[j] = *(const f32x4*)(Wo + (size_t)(n0 + row) * EMBED + k0 + 32 + kk);
      }
    }

    Frag a[2], b[4];
    #pragma unroll
    for (int mt = 0; mt < 2; ++mt) {
      int row = wm * 32 + mt * 16 + (lane & 15);
      a[mt].u[0] = *(const u32x4*)&sA[row * P_LDA + kh0];
      a[mt].u[1] = *(const u32x4*)&sA[row * P_LDA + kh0 + 16];
    }
    #pragma unroll
    for (int nt = 0; nt < 4; ++nt) {
      int col = wn * 64 + nt * 16 + (lane & 15);
      b[nt].u[0] = *(const u32x4*)&sB[col * P_LDA + kh0];
      b[nt].u[1] = *(const u32x4*)&sB[col * P_LDA + kh0 + 16];
    }
    #pragma unroll
    for (int mt = 0; mt < 2; ++mt)
      #pragma unroll
      for (int nt = 0; nt < 4; ++nt)
        acc[mt][nt] = wmma_bf16(a[mt], b[nt], acc[mt][nt]);
    __syncthreads();
  }

  #pragma unroll
  for (int mt = 0; mt < 2; ++mt) {
    int rbase = m0 + wm * 32 + mt * 16 + ((lane >= 16) ? 8 : 0);
    #pragma unroll
    for (int nt = 0; nt < 4; ++nt) {
      int n = n0 + wn * 64 + nt * 16 + (lane & 15);
      float bb = bo[n];
      #pragma unroll
      for (int r = 0; r < 8; ++r)
        out[(size_t)(rbase + r) * EMBED + n] = acc[mt][nt][r] + bb;
    }
  }
}

// ============================================================================
extern "C" void kernel_launch(void* const* d_in, const int* in_sizes, int n_in,
                              void* d_out, int out_size, void* d_ws, size_t ws_size,
                              hipStream_t stream) {
  const float* q  = (const float*)d_in[0];
  const float* k  = (const float*)d_in[1];
  const float* v  = (const float*)d_in[2];
  const float* Wq = (const float*)d_in[3];
  const float* bq = (const float*)d_in[4];
  const float* Wk = (const float*)d_in[5];
  const float* bk = (const float*)d_in[6];
  const float* Wv = (const float*)d_in[7];
  const float* bv = (const float*)d_in[8];
  const float* Wo = (const float*)d_in[9];
  const float* bo = (const float*)d_in[10];
  float* out = (float*)d_out;

  char* ws = (char*)d_ws;
  const size_t SZ = (size_t)NBATCH * HEADS * SEQ * HDIM * sizeof(unsigned short); // 16 MiB
  unsigned short* qws  = (unsigned short*)(ws);
  unsigned short* kws  = (unsigned short*)(ws + SZ);
  unsigned short* vtws = (unsigned short*)(ws + 2 * SZ);
  unsigned short* aws  = (unsigned short*)(ws + 3 * SZ);

  proj_gemm<<<dim3(EMBED / 128, MTOTAL / 128, 3), 256, 0, stream>>>(
      q, k, v, Wq, Wk, Wv, bq, bk, bv, qws, kws, vtws);
  attn_kernel<<<dim3(SEQ / 256, HEADS, NBATCH), 256, 0, stream>>>(
      qws, kws, vtws, aws);
  outproj_gemm<<<dim3(EMBED / 128, MTOTAL / 128), 256, 0, stream>>>(
      aws, Wo, bo, out);
}
